// RobustFill_36953898615026
// MI455X (gfx1250) — matrix operand order repaired and checked by reading, not compiled, not run on hardware
//
#include <hip/hip_runtime.h>
#include <cstddef>

// ---------------------------------------------------------------------------
// RobustFill (3x 2-layer LSTM, H=2048, batch=1) for MI455X / gfx1250.
// Strategy:
//  * One-time repack of all weights fp32 -> bf16 in WMMA A-fragment layout.
//    Per-phase bf16 weight set (~104-134MB) fits the 192MB L2, so the
//    sequential recurrence streams weights from L2 instead of HBM.
//  * Mat-vec via v_wmma_f32_16x16x32_bf16 with the vector replicated across
//    the 16 B-columns (column 0 of D used).  Bandwidth-bound => the 16x
//    FLOP waste is free.
//  * 1 block = 4 waves = 4 gates of one 16-row strip; gate activations and
//    c/h update fused in-kernel through LDS.  h double-buffered.
//  * Final projection done as a real WMMA GEMM [64x2048]x[2048x1024]+bias.
// ---------------------------------------------------------------------------

typedef __attribute__((ext_vector_type(16))) __bf16        v16bf;
typedef __attribute__((ext_vector_type(8)))  float         v8f;
typedef __attribute__((ext_vector_type(4)))  unsigned int  u32x4;

union Frag16 { v16bf bf; u32x4 q[2]; };

#define H_DIM  2048
#define S_DIM  256
#define P_DIM  1024
#define R4H    8192      // 4*H
#define T_IN   128
#define T_OUT  128
#define P_LEN  64

__device__ __forceinline__ unsigned short f32_to_bf16(float f) {
  unsigned int u = __float_as_uint(f);
  unsigned int lsb = (u >> 16) & 1u;
  u += 0x7fffu + lsb;                 // round-to-nearest-even
  return (unsigned short)(u >> 16);
}

__device__ __forceinline__ float sigm(float x) { return 1.0f / (1.0f + expf(-x)); }

// ---------------------------------------------------------------------------
// fp32 -> bf16 elementwise convert (row-major copy)
// ---------------------------------------------------------------------------
__global__ void cvt_bf16_kernel(const float* __restrict__ in,
                                unsigned short* __restrict__ out, int n) {
  int i = blockIdx.x * blockDim.x + threadIdx.x;
  if (i < n) out[i] = f32_to_bf16(in[i]);
}

// ---------------------------------------------------------------------------
// Pack a row-major [R4H x K] fp32 matrix into bf16 WMMA A-fragments.
// Fragment (strip s, kstep t): 32 lanes x 32B, lane layout per CDNA5 ISA
// 16-bit A 16x32 table:  lane(half,m) run r holds K = t*32 + half*8 + r*16 +e
// Flat: elem_offset = (((s*nK + t)*32 + lane)*2 + run)*8
// ---------------------------------------------------------------------------
__global__ void pack_a_kernel(const float* __restrict__ W,
                              unsigned short* __restrict__ out,
                              int K, int total) {
  int idx = blockIdx.x * blockDim.x + threadIdx.x;
  if (idx >= total) return;
  int run  = idx & 1;
  int lane = (idx >> 1) & 31;
  int rest = idx >> 6;
  int nk = K >> 5;
  int t = rest % nk;
  int s = rest / nk;
  int half = lane >> 4, m = lane & 15;
  int kb = (t << 5) + half * 8 + run * 16;
  const float* src = W + (size_t)(s * 16 + m) * K + kb;
  unsigned short* dst = out + (size_t)idx * 8;
#pragma unroll
  for (int e = 0; e < 8; ++e) dst[e] = f32_to_bf16(src[e]);
}

// ---------------------------------------------------------------------------
// Fused LSTM cell step:  z = W@x + U@h + b ; gate update ; h_out, c in/out.
// grid  = H/16 = 128 blocks  (one 16-row strip of each gate)
// block = 128 threads = 4 waves, wave g computes rows [g*H + 16*bx, +16)
// ---------------------------------------------------------------------------
__global__ __launch_bounds__(128)
void lstm_cell_kernel(const unsigned short* __restrict__ Wp, int KW,
                      const unsigned short* __restrict__ Up,
                      const unsigned short* __restrict__ x,    // bf16 [KW]
                      const unsigned short* __restrict__ h_in, // bf16 [H]
                      const float* __restrict__ b,             // fp32 [4H]
                      float* __restrict__ c,                   // fp32 [H] in/out
                      unsigned short* __restrict__ h_out,      // bf16 [H]
                      unsigned short* __restrict__ hbuf_row)   // optional bf16 [H]
{
  const int lane = threadIdx.x & 31;
  const int gate = threadIdx.x >> 5;
  const int half = lane >> 4;
  const int strip = gate * (H_DIM / 16) + blockIdx.x;   // strip over 4H rows

  v8f acc = {};

  // ---- W @ x -------------------------------------------------------------
  {
    const int nk = KW >> 5;
    const u32x4* abase =
        reinterpret_cast<const u32x4*>(Wp) + ((size_t)strip * nk * 32 + lane) * 2;
    for (int t = 0; t < nk; ++t) {
      Frag16 a, bb;
      const u32x4* ap = abase + (size_t)t * 64;   // 32 lanes * 2 quads
      a.q[0] = ap[0];
      a.q[1] = ap[1];
      const u32x4* bp = reinterpret_cast<const u32x4*>(x + (t << 5) + half * 8);
      bb.q[0] = bp[0];
      bb.q[1] = bp[2];
      acc = __builtin_amdgcn_wmma_f32_16x16x32_bf16(false, a.bf, false, bb.bf,
                                                    (short)0, acc, false, false);
    }
  }
  // ---- U @ h -------------------------------------------------------------
  {
    const int nk = H_DIM >> 5;   // 64
    const u32x4* abase =
        reinterpret_cast<const u32x4*>(Up) + ((size_t)strip * nk * 32 + lane) * 2;
    for (int t = 0; t < nk; ++t) {
      Frag16 a, bb;
      const u32x4* ap = abase + (size_t)t * 64;
      a.q[0] = ap[0];
      a.q[1] = ap[1];
      const u32x4* bp = reinterpret_cast<const u32x4*>(h_in + (t << 5) + half * 8);
      bb.q[0] = bp[0];
      bb.q[1] = bp[2];
      acc = __builtin_amdgcn_wmma_f32_16x16x32_bf16(false, a.bf, false, bb.bf,
                                                    (short)0, acc, false, false);
    }
  }

  // Column 0 of D lives in lanes 0 (M=0..7) and 16 (M=8..15).
  __shared__ float zsh[4][16];
  if (lane == 0) {
#pragma unroll
    for (int r = 0; r < 8; ++r) zsh[gate][r] = acc[r];
  } else if (lane == 16) {
#pragma unroll
    for (int r = 0; r < 8; ++r) zsh[gate][8 + r] = acc[r];
  }
  __syncthreads();

  const int tid = threadIdx.x;
  if (tid < 16) {
    const int row = blockIdx.x * 16 + tid;
    float zi = zsh[0][tid] + b[0 * H_DIM + row];
    float zf = zsh[1][tid] + b[1 * H_DIM + row];
    float zg = zsh[2][tid] + b[2 * H_DIM + row];
    float zo = zsh[3][tid] + b[3 * H_DIM + row];
    float co = c[row];
    float cn = sigm(zf) * co + sigm(zi) * tanhf(zg);
    float hn = sigm(zo) * tanhf(cn);
    c[row] = cn;
    unsigned short hb = f32_to_bf16(hn);
    h_out[row] = hb;
    if (hbuf_row) hbuf_row[row] = hb;
  }
}

// ---------------------------------------------------------------------------
// Final projection: out[64,1024] = Hbuf[64,2048] @ lin_W^T + lin_b
// grid = (16,4), block = 128 (4 waves); wave -> one 16x16 output tile.
// ---------------------------------------------------------------------------
__global__ __launch_bounds__(128)
void out_gemm_kernel(const unsigned short* __restrict__ Hbuf, // bf16 [64,2048]
                     const unsigned short* __restrict__ Wb,   // bf16 [1024,2048]
                     const float* __restrict__ bias,          // fp32 [1024]
                     float* __restrict__ out)                 // fp32 [64,1024]
{
  const int lane  = threadIdx.x & 31;
  const int wave  = threadIdx.x >> 5;
  const int half  = lane >> 4;
  const int m     = lane & 15;
  const int mtile = blockIdx.y;
  const int ntile = blockIdx.x * 4 + wave;
  const int arow  = mtile * 16 + m;       // A row (timestep)
  const int bcol  = ntile * 16 + m;       // B column (output unit)

  v8f acc = {};
  for (int t = 0; t < (H_DIM >> 5); ++t) {
    const int kb = t << 5;
    Frag16 a, bb;
    const u32x4* ap0 =
        reinterpret_cast<const u32x4*>(Hbuf + (size_t)arow * H_DIM + kb + half * 8);
    const u32x4* ap1 =
        reinterpret_cast<const u32x4*>(Hbuf + (size_t)arow * H_DIM + kb + 16 + half * 8);
    a.q[0] = ap0[0];
    a.q[1] = ap1[0];
    const u32x4* bp =
        reinterpret_cast<const u32x4*>(Wb + (size_t)bcol * H_DIM + kb + half * 8);
    bb.q[0] = bp[0];
    bb.q[1] = bp[2];
    acc = __builtin_amdgcn_wmma_f32_16x16x32_bf16(false, a.bf, false, bb.bf,
                                                  (short)0, acc, false, false);
  }
#pragma unroll
  for (int r = 0; r < 8; ++r) {
    const int orow = mtile * 16 + half * 8 + r;
    const int ocol = ntile * 16 + m;
    out[(size_t)orow * P_DIM + ocol] = acc[r] + bias[ocol];
  }
}

// ---------------------------------------------------------------------------
// Host side
// ---------------------------------------------------------------------------
static inline size_t carve(size_t& off, size_t bytes) {
  off = (off + 255) & ~(size_t)255;
  size_t p = off;
  off += bytes;
  return p;
}

extern "C" void kernel_launch(void* const* d_in, const int* in_sizes, int n_in,
                              void* d_out, int out_size, void* d_ws, size_t ws_size,
                              hipStream_t stream) {
  (void)in_sizes; (void)n_in; (void)out_size;

  const float* in_seq  = (const float*)d_in[0];
  const float* out_seq = (const float*)d_in[1];
  const float* iW0 = (const float*)d_in[2];
  const float* iU0 = (const float*)d_in[3];
  const float* ib0 = (const float*)d_in[4];
  const float* iW1 = (const float*)d_in[5];
  const float* iU1 = (const float*)d_in[6];
  const float* ib1 = (const float*)d_in[7];
  const float* oW0 = (const float*)d_in[8];
  const float* oU0 = (const float*)d_in[9];
  const float* ob0 = (const float*)d_in[10];
  const float* oW1 = (const float*)d_in[11];
  const float* oU1 = (const float*)d_in[12];
  const float* ob1 = (const float*)d_in[13];
  const float* pW0 = (const float*)d_in[14];
  const float* pU0 = (const float*)d_in[15];
  const float* pb0 = (const float*)d_in[16];
  const float* pW1 = (const float*)d_in[17];
  const float* pU1 = (const float*)d_in[18];
  const float* pb1 = (const float*)d_in[19];
  const float* linW = (const float*)d_in[20];
  const float* linB = (const float*)d_in[21];

  // ---- workspace layout --------------------------------------------------
  const size_t BYT_W_S = (size_t)R4H * S_DIM * 2;   // 4 MB  (bf16 [4H x S])
  const size_t BYT_W_H = (size_t)R4H * H_DIM * 2;   // 32 MB (bf16 [4H x H])
  size_t off = 0;
  size_t o_iW0p = carve(off, BYT_W_S), o_iU0p = carve(off, BYT_W_H);
  size_t o_iW1p = carve(off, BYT_W_H), o_iU1p = carve(off, BYT_W_H);
  size_t o_oW0p = carve(off, BYT_W_S), o_oU0p = carve(off, BYT_W_H);
  size_t o_oW1p = carve(off, BYT_W_H), o_oU1p = carve(off, BYT_W_H);
  size_t o_pW0p = carve(off, BYT_W_H), o_pU0p = carve(off, BYT_W_H);
  size_t o_pW1p = carve(off, BYT_W_H), o_pU1p = carve(off, BYT_W_H);
  size_t o_linWb = carve(off, (size_t)P_DIM * H_DIM * 2);
  size_t o_xin   = carve(off, (size_t)T_IN * S_DIM * 2);
  size_t o_xout  = carve(off, (size_t)T_OUT * S_DIM * 2);
  size_t o_h0A = carve(off, H_DIM * 2), o_h0B = carve(off, H_DIM * 2);
  size_t o_h1A = carve(off, H_DIM * 2), o_h1B = carve(off, H_DIM * 2);
  size_t o_prev = carve(off, H_DIM * 2);
  size_t o_c0 = carve(off, H_DIM * 4), o_c1 = carve(off, H_DIM * 4);
  size_t o_hbuf = carve(off, (size_t)P_LEN * H_DIM * 2);
  if (off > ws_size) return;   // workspace too small: deterministic no-op

  char* ws = (char*)d_ws;
  auto US = [&](size_t o) { return (unsigned short*)(ws + o); };
  auto FP = [&](size_t o) { return (float*)(ws + o); };

  // ---- one-time prep: convert + pack ------------------------------------
  {
    int n = T_IN * S_DIM;
    cvt_bf16_kernel<<<(n + 255) / 256, 256, 0, stream>>>(in_seq, US(o_xin), n);
    n = T_OUT * S_DIM;
    cvt_bf16_kernel<<<(n + 255) / 256, 256, 0, stream>>>(out_seq, US(o_xout), n);
    n = P_DIM * H_DIM;
    cvt_bf16_kernel<<<(n + 255) / 256, 256, 0, stream>>>(linW, US(o_linWb), n);
  }
  auto pack = [&](const float* src, size_t dst, int K) {
    int total = (R4H / 16) * (K / 32) * 64;
    pack_a_kernel<<<(total + 255) / 256, 256, 0, stream>>>(src, US(dst), K, total);
  };
  pack(iW0, o_iW0p, S_DIM); pack(iU0, o_iU0p, H_DIM);
  pack(iW1, o_iW1p, H_DIM); pack(iU1, o_iU1p, H_DIM);
  pack(oW0, o_oW0p, S_DIM); pack(oU0, o_oU0p, H_DIM);
  pack(oW1, o_oW1p, H_DIM); pack(oU1, o_oU1p, H_DIM);
  pack(pW0, o_pW0p, H_DIM); pack(pU0, o_pU0p, H_DIM);
  pack(pW1, o_pW1p, H_DIM); pack(pU1, o_pU1p, H_DIM);

  // ---- zero state every call (harness does not re-poison) ----------------
  hipMemsetAsync(US(o_h0A), 0, H_DIM * 2, stream);
  hipMemsetAsync(US(o_h1A), 0, H_DIM * 2, stream);
  hipMemsetAsync(FP(o_c0), 0, H_DIM * 4, stream);
  hipMemsetAsync(FP(o_c1), 0, H_DIM * 4, stream);

  unsigned short *h0 = US(o_h0A), *h0n = US(o_h0B);
  unsigned short *h1 = US(o_h1A), *h1n = US(o_h1B);
  const dim3 cgrid(H_DIM / 16), cblk(128);

  auto step2 = [&](const unsigned short* W0p, int KW0, const unsigned short* U0p,
                   const float* b0, const unsigned short* W1p,
                   const unsigned short* U1p, const float* b1,
                   const unsigned short* x, unsigned short* hbuf_row) {
    lstm_cell_kernel<<<cgrid, cblk, 0, stream>>>(W0p, KW0, U0p, x, h0, b0,
                                                 FP(o_c0), h0n, nullptr);
    lstm_cell_kernel<<<cgrid, cblk, 0, stream>>>(W1p, H_DIM, U1p, h0n, h1, b1,
                                                 FP(o_c1), h1n, hbuf_row);
    unsigned short* t0 = h0; h0 = h0n; h0n = t0;
    unsigned short* t1 = h1; h1 = h1n; h1n = t1;
  };

  // ---- input encode ------------------------------------------------------
  for (int t = 0; t < T_IN; ++t)
    step2(US(o_iW0p), S_DIM, US(o_iU0p), ib0, US(o_iW1p), US(o_iU1p), ib1,
          US(o_xin) + (size_t)t * S_DIM, nullptr);
  // ---- output encode -----------------------------------------------------
  for (int t = 0; t < T_OUT; ++t)
    step2(US(o_oW0p), S_DIM, US(o_oU0p), ob0, US(o_oW1p), US(o_oU1p), ob1,
          US(o_xout) + (size_t)t * S_DIM, nullptr);

  // ---- program LSTM: constant input = snapshot of h1 ---------------------
  hipMemcpyAsync(US(o_prev), h1, H_DIM * 2, hipMemcpyDeviceToDevice, stream);
  for (int t = 0; t < P_LEN; ++t)
    step2(US(o_pW0p), H_DIM, US(o_pU0p), pb0, US(o_pW1p), US(o_pU1p), pb1,
          US(o_prev), US(o_hbuf) + (size_t)t * H_DIM);

  // ---- final projection GEMM (fused bias) --------------------------------
  out_gemm_kernel<<<dim3(P_DIM / 64, P_LEN / 16), 128, 0, stream>>>(
      US(o_hbuf), US(o_linWb), linB, (float*)d_out);
}
